// DAEncoderLayer_89309549953424
// MI455X (gfx1250) — compile-verified, matrix operand order checked by reading, not run on hardware
//
#include <hip/hip_runtime.h>

// ---------------------------------------------------------------------------
// Deformable attention encoder layer for MI455X (gfx1250, wave32).
// All dense GEMMs run on v_wmma_f32_16x16x32_bf16; activations feeding GEMMs
// are stored bf16; the value map is stored bf16 in (B,NH,LQ,DH) layout so the
// bilinear gather stays inside the 192MB L2. Bandwidth-bound design.
// ---------------------------------------------------------------------------

typedef __attribute__((ext_vector_type(16))) __bf16 v16bf;
typedef __attribute__((ext_vector_type(8)))  __bf16 v8bf;
typedef __attribute__((ext_vector_type(8)))  float  v8f;

#define NROWS 65536   // B*LQ = 16*4096
#define CDIM  256

__device__ __forceinline__ __bf16 f2bf(float f) {
  union { float f; unsigned u; } v; v.f = f;
  unsigned r = v.u + 0x7FFFu + ((v.u >> 16) & 1u);   // round-to-nearest-even
  union { unsigned short s; __bf16 b; } o;
  o.s = (unsigned short)(r >> 16);
  return o.b;
}

// ---------------------------------------------------------------------------
// prep: qbf = bf16(src + pos), sbf = bf16(src)
// ---------------------------------------------------------------------------
__global__ __launch_bounds__(256)
void prep_k(const float* __restrict__ src, const float* __restrict__ pos,
            __bf16* __restrict__ qbf, __bf16* __restrict__ sbf) {
  size_t i = (size_t)blockIdx.x * 256 + threadIdx.x;
  float s = src[i];
  qbf[i] = f2bf(s + pos[i]);
  sbf[i] = f2bf(s);
}

// ---------------------------------------------------------------------------
// pack: fp32 weight W[K][N] (row-major) -> per-lane WMMA B-operand order.
// packed element i = (kt*(N/16)+nt)*512 + lane*16 + e holds W[k][n] with
//   k = kt*32 + e + (e>=8 ? 8 : 0) + (lane>=16 ? 8 : 0),  n = nt*16 + lane%16
// matching the 16-bit 32x16 B-matrix VGPR layout (ISA 7.12.2).
// ---------------------------------------------------------------------------
__global__ __launch_bounds__(256)
void pack_k(const float* __restrict__ W, __bf16* __restrict__ out, int K, int N) {
  int i = blockIdx.x * 256 + threadIdx.x;
  if (i >= K * N) return;
  int e    = i & 15;
  int lane = (i >> 4) & 31;
  int t    = i >> 9;
  int ntiles = N >> 4;
  int nt = t % ntiles;
  int kt = t / ntiles;
  int k = kt * 32 + e + ((e & 8) ? 8 : 0) + ((lane & 16) ? 8 : 0);
  int n = nt * 16 + (lane & 15);
  out[i] = f2bf(W[(size_t)k * N + n]);
}

// ---------------------------------------------------------------------------
// GEMM: C[M=65536, N] = A[M,K](bf16 row-major) x Bp(packed bf16) + bias
// One wave -> 16x32 output (2 accumulators); 8 waves/block -> 128 rows.
// OUTMODE: 0 = fp32 row-major, 1 = bf16 row-major, 2 = bf16 (B,NH,LQ,DH)
// ---------------------------------------------------------------------------
template<int K, bool RELU, int OUTMODE>
__global__ __launch_bounds__(256)
void gemm_bf16_k(const __bf16* __restrict__ A, const __bf16* __restrict__ Bp,
                 const float* __restrict__ bias, float* __restrict__ Cf,
                 __bf16* __restrict__ Cb, int N) {
  const int lane = threadIdx.x & 31;
  const int wave = threadIdx.x >> 5;
  const int m0   = (blockIdx.y * 8 + wave) * 16;
  const int nt0  = blockIdx.x * 2;
  const int ntiles = N >> 4;
  const int row  = m0 + (lane & 15);
  const int koff = (lane & 16) ? 8 : 0;
  const __bf16* arow = A + (size_t)row * K + koff;

  v8f acc0 = {}; v8f acc1 = {};
#pragma unroll
  for (int kt = 0; kt < K / 32; ++kt) {
    v16bf a;
    v8bf lo = *(const v8bf*)(arow + kt * 32);        // K = k0+{0..7}  (+8 if hi lane)
    v8bf hi = *(const v8bf*)(arow + kt * 32 + 16);   // K = k0+{16..23}(+8 if hi lane)
#pragma unroll
    for (int j = 0; j < 8; ++j) { a[j] = lo[j]; a[j + 8] = hi[j]; }
    v16bf b0 = *(const v16bf*)(Bp + (((size_t)(kt * ntiles + nt0)    ) * 32 + lane) * 16);
    v16bf b1 = *(const v16bf*)(Bp + (((size_t)(kt * ntiles + nt0 + 1)) * 32 + lane) * 16);
    acc0 = __builtin_amdgcn_wmma_f32_16x16x32_bf16(false, a, false, b0, (short)0, acc0, false, false);
    acc1 = __builtin_amdgcn_wmma_f32_16x16x32_bf16(false, a, false, b1, (short)0, acc1, false, false);
  }

  const int col0  = nt0 * 16 + (lane & 15);
  const int rbase = m0 + ((lane & 16) ? 8 : 0);
  const float bb0 = bias[col0];
  const float bb1 = bias[col0 + 16];
#pragma unroll
  for (int i = 0; i < 8; ++i) {
    int r = rbase + i;
    float v0 = acc0[i] + bb0;
    float v1 = acc1[i] + bb1;
    if (RELU) { v0 = fmaxf(v0, 0.f); v1 = fmaxf(v1, 0.f); }
    if (OUTMODE == 0) {
      Cf[(size_t)r * N + col0]      = v0;
      Cf[(size_t)r * N + col0 + 16] = v1;
    } else if (OUTMODE == 1) {
      Cb[(size_t)r * N + col0]      = f2bf(v0);
      Cb[(size_t)r * N + col0 + 16] = f2bf(v1);
    } else {  // value map layout: (B, NH=8, LQ=4096, DH=32)
      int b = r >> 12, lq = r & 4095;
      int c0 = col0, c1 = col0 + 16;
      Cb[(((size_t)b * 8 + (c0 >> 5)) * 4096 + lq) * 32 + (c0 & 31)] = f2bf(v0);
      Cb[(((size_t)b * 8 + (c1 >> 5)) * 4096 + lq) * 32 + (c1 & 31)] = f2bf(v1);
    }
  }
}

// ---------------------------------------------------------------------------
// Sampling + softmax + point-weighted accumulation.
// One wave per (query, head); lane = channel d (DH=32 == wave32). The 33.5MB
// bf16 value map is L2-resident, so the 16 corner gathers/query are cheap.
// ---------------------------------------------------------------------------
__global__ __launch_bounds__(256)
void sample_attn_k(const float* __restrict__ proj_off, const float* __restrict__ proj_attn,
                   const float* __restrict__ refp, const __bf16* __restrict__ val,
                   __bf16* __restrict__ attn_bf) {
  int gid  = blockIdx.x * 8 + (threadIdx.x >> 5);
  int lane = threadIdx.x & 31;
  int h    = gid & 7;
  size_t bq = (size_t)(gid >> 3);
  int b = (int)(bq >> 12);

  const float* at = proj_attn + bq * 32 + h * 4;
  float l0 = at[0], l1 = at[1], l2 = at[2], l3 = at[3];
  float mx = fmaxf(fmaxf(l0, l1), fmaxf(l2, l3));
  float e0 = __expf(l0 - mx), e1 = __expf(l1 - mx),
        e2 = __expf(l2 - mx), e3 = __expf(l3 - mx);
  float inv = 1.f / (e0 + e1 + e2 + e3);
  float aw[4] = { e0 * inv, e1 * inv, e2 * inv, e3 * inv };

  const float* of = proj_off + bq * 64 + h * 8;
  float rx = refp[bq * 2], ry = refp[bq * 2 + 1];
  const __bf16* vb = val + ((size_t)b * 8 + h) * 4096 * 32 + lane;

  float acc = 0.f;
#pragma unroll
  for (int p = 0; p < 4; ++p) {
    float x = (rx + of[2 * p]     * (1.f / 64.f)) * 64.f - 0.5f;
    float y = (ry + of[2 * p + 1] * (1.f / 64.f)) * 64.f - 0.5f;
    float xf = floorf(x), yf = floorf(y);
    float wx = x - xf, wy = y - yf;
    int x0 = (int)xf, y0 = (int)yf;
    float s[4];
    int xs[2] = { x0, x0 + 1 }, ys[2] = { y0, y0 + 1 };
#pragma unroll
    for (int cy = 0; cy < 2; ++cy)
#pragma unroll
      for (int cx = 0; cx < 2; ++cx) {
        int ix = xs[cx], iy = ys[cy];
        bool ok = (ix >= 0) && (ix < 64) && (iy >= 0) && (iy < 64);
        int cxi = ix < 0 ? 0 : (ix > 63 ? 63 : ix);
        int cyi = iy < 0 ? 0 : (iy > 63 ? 63 : iy);
        float sv = (float)vb[(size_t)(cyi * 64 + cxi) * 32];
        s[cy * 2 + cx] = ok ? sv : 0.f;   // zeroed sample == masked weight
      }
    float bi = s[0] * (1 - wx) * (1 - wy) + s[1] * wx * (1 - wy)
             + s[2] * (1 - wx) * wy       + s[3] * wx * wy;
    acc += aw[p] * bi;
  }
  attn_bf[bq * 256 + h * 32 + lane] = f2bf(acc);
}

// ---------------------------------------------------------------------------
// LayerNorm over C=256: one wave per row, 8 elems/lane, shfl_xor reductions.
// Writes fp32 (for residual / final out) and optionally bf16 (GEMM feed).
// ---------------------------------------------------------------------------
__global__ __launch_bounds__(256)
void ln_k(const float* __restrict__ resid, const float* __restrict__ y,
          const float* __restrict__ g, const float* __restrict__ be,
          float* __restrict__ xf, __bf16* __restrict__ xb) {
  size_t row = (size_t)blockIdx.x * 8 + (threadIdx.x >> 5);
  int lane = threadIdx.x & 31;
  const float* rr = resid + row * 256 + lane * 8;
  const float* yy = y     + row * 256 + lane * 8;
  float v[8]; float s = 0.f;
#pragma unroll
  for (int i = 0; i < 8; ++i) { v[i] = rr[i] + yy[i]; s += v[i]; }
#pragma unroll
  for (int o = 16; o > 0; o >>= 1) s += __shfl_xor(s, o, 32);
  float mean = s * (1.f / 256.f);
  float q = 0.f;
#pragma unroll
  for (int i = 0; i < 8; ++i) { float c = v[i] - mean; q += c * c; }
#pragma unroll
  for (int o = 16; o > 0; o >>= 1) q += __shfl_xor(q, o, 32);
  float rinv = rsqrtf(q * (1.f / 256.f) + 1e-5f);
#pragma unroll
  for (int i = 0; i < 8; ++i) {
    int col = lane * 8 + i;
    float o = (v[i] - mean) * rinv * g[col] + be[col];
    xf[row * 256 + col] = o;
    if (xb) xb[row * 256 + col] = f2bf(o);
  }
}

// ---------------------------------------------------------------------------
extern "C" void kernel_launch(void* const* d_in, const int* in_sizes, int n_in,
                              void* d_out, int out_size, void* d_ws, size_t ws_size,
                              hipStream_t stream) {
  (void)in_sizes; (void)n_in; (void)out_size; (void)ws_size;
  const float* src    = (const float*)d_in[0];
  const float* pos    = (const float*)d_in[1];
  const float* refp   = (const float*)d_in[2];
  // d_in[3] spatial_shapes (int64) unused: HS=WS=64 hardcoded
  const float* W_off  = (const float*)d_in[4];
  const float* b_off  = (const float*)d_in[5];
  const float* W_attn = (const float*)d_in[6];
  const float* b_attn = (const float*)d_in[7];
  const float* W_val  = (const float*)d_in[8];
  const float* b_val  = (const float*)d_in[9];
  const float* W_out  = (const float*)d_in[10];
  const float* b_out  = (const float*)d_in[11];
  const float* ln1g   = (const float*)d_in[12];
  const float* ln1b   = (const float*)d_in[13];
  const float* W1     = (const float*)d_in[14];
  const float* b1     = (const float*)d_in[15];
  const float* W2     = (const float*)d_in[16];
  const float* b2     = (const float*)d_in[17];
  const float* ln2g   = (const float*)d_in[18];
  const float* ln2b   = (const float*)d_in[19];

  // Workspace layout (bytes). h_bf aliases the dead early-phase region.
  char* ws = (char*)d_ws;
  __bf16* qbf       = (__bf16*)(ws + 0);            // 33,554,432
  __bf16* sbf       = (__bf16*)(ws + 33554432);     // 33,554,432
  float*  proj_off  = (float*) (ws + 67108864);     // 16,777,216
  float*  proj_attn = (float*) (ws + 83886080);     //  8,388,608
  __bf16* val_bf    = (__bf16*)(ws + 92274688);     // 33,554,432
  __bf16* attn_bf   = (__bf16*)(ws + 125829120);    // 33,554,432
  __bf16* h_bf      = (__bf16*)(ws + 0);            // 134,217,728 (alias)
  float*  tmp       = (float*) (ws + 159383552);    // 67,108,864
  float*  x1f       = (float*) (ws + 226492416);    // 67,108,864
  __bf16* x1bf      = (__bf16*)(ws + 293601280);    // 33,554,432
  char*   wb        = ws + 327155712;
  __bf16* Wp_off  = (__bf16*)(wb);                  //  32,768
  __bf16* Wp_attn = (__bf16*)(wb + 32768);          //  16,384
  __bf16* Wp_val  = (__bf16*)(wb + 49152);          // 131,072
  __bf16* Wp_out  = (__bf16*)(wb + 180224);         // 131,072
  __bf16* Wp1     = (__bf16*)(wb + 311296);         // 524,288
  __bf16* Wp2     = (__bf16*)(wb + 835584);         // 524,288
  float* outp = (float*)d_out;

  dim3 blk(256);
  // 1) activations -> bf16
  prep_k<<<NROWS, blk, 0, stream>>>(src, pos, qbf, sbf);
  // 2) weights -> WMMA-packed bf16
  pack_k<<<(256 * 64   + 255) / 256, blk, 0, stream>>>(W_off,  Wp_off,  256, 64);
  pack_k<<<(256 * 32   + 255) / 256, blk, 0, stream>>>(W_attn, Wp_attn, 256, 32);
  pack_k<<<(256 * 256  + 255) / 256, blk, 0, stream>>>(W_val,  Wp_val,  256, 256);
  pack_k<<<(256 * 256  + 255) / 256, blk, 0, stream>>>(W_out,  Wp_out,  256, 256);
  pack_k<<<(256 * 1024 + 255) / 256, blk, 0, stream>>>(W1,     Wp1,     256, 1024);
  pack_k<<<(1024 * 256 + 255) / 256, blk, 0, stream>>>(W2,     Wp2,     1024, 256);
  // 3) projection GEMMs (WMMA bf16)
  gemm_bf16_k<256, false, 0><<<dim3(2, 512),  blk, 0, stream>>>(qbf, Wp_off,  b_off,  proj_off,  nullptr, 64);
  gemm_bf16_k<256, false, 0><<<dim3(1, 512),  blk, 0, stream>>>(qbf, Wp_attn, b_attn, proj_attn, nullptr, 32);
  gemm_bf16_k<256, false, 2><<<dim3(8, 512),  blk, 0, stream>>>(sbf, Wp_val,  b_val,  nullptr,   val_bf,  256);
  // 4) softmax + bilinear sampling + point accumulation
  sample_attn_k<<<NROWS, blk, 0, stream>>>(proj_off, proj_attn, refp, val_bf, attn_bf);
  // 5) output projection + LN1
  gemm_bf16_k<256, false, 0><<<dim3(8, 512),  blk, 0, stream>>>(attn_bf, Wp_out, b_out, tmp, nullptr, 256);
  ln_k<<<NROWS / 8, blk, 0, stream>>>(src, tmp, ln1g, ln1b, x1f, x1bf);
  // 6) FFN (ReLU fused) + LN2
  gemm_bf16_k<256, true,  1><<<dim3(32, 512), blk, 0, stream>>>(x1bf, Wp1, b1, nullptr, h_bf, 1024);
  gemm_bf16_k<1024, false, 0><<<dim3(8, 512), blk, 0, stream>>>(h_bf, Wp2, b2, tmp, nullptr, 256);
  ln_k<<<NROWS / 8, blk, 0, stream>>>(x1f, tmp, ln2g, ln2b, outp, nullptr);
}
